// Decoder_42202348650840
// MI455X (gfx1250) — matrix operand order repaired and checked by reading, not compiled, not run on hardware
//
#include <hip/hip_runtime.h>
#include <hip/hip_bf16.h>
#include <math.h>

// ---------------- problem constants (from reference) ----------------
#define B_   32
#define T_   100
#define TE_  100
#define D_   1024
#define L_   6
#define H_   8
#define DH_  64
#define V_   32000
#define M_   (B_ * T_)      // 3200 rows of activations
#define HD_  (H_ * DH_)     // 512
#define FFD_ (4 * D_)       // 4096

// ---------------- CDNA5 types ----------------
typedef __attribute__((ext_vector_type(16))) _Float16 v16h;
typedef __attribute__((ext_vector_type(8)))  float    v8f;

__device__ __forceinline__ v8f vzero8() {
    v8f z = {0.f, 0.f, 0.f, 0.f, 0.f, 0.f, 0.f, 0.f};
    return z;
}

// ---------------- async global->LDS staging (ASYNCcnt path) ----------------
// global_load_async_to_lds_b128: vdst = LDS byte address, vaddr = 64-bit global.
#define USE_ASYNC_LDS 1
#define USE_NT_HINT   1

__device__ __forceinline__ void g2l_b128(void* lds, const void* g) {
#if USE_ASYNC_LDS
    unsigned int lo = (unsigned int)(unsigned long long)lds;
    asm volatile("global_load_async_to_lds_b128 %0, %1, off"
                 :: "v"(lo), "v"(g) : "memory");
#else
    *(float4*)lds = *(const float4*)g;
#endif
}

// weights are streamed exactly once -> non-temporal, don't pollute L2
__device__ __forceinline__ void g2l_b128_nt(void* lds, const void* g) {
#if USE_ASYNC_LDS && USE_NT_HINT
    unsigned int lo = (unsigned int)(unsigned long long)lds;
    asm volatile("global_load_async_to_lds_b128 %0, %1, off th:TH_LOAD_NT"
                 :: "v"(lo), "v"(g) : "memory");
#else
    g2l_b128(lds, g);
#endif
}

// wait until at most N async transfers of THIS wave are outstanding
__device__ __forceinline__ void async_wait_le6() {
#if USE_ASYNC_LDS
    asm volatile("s_wait_asynccnt 6" ::: "memory");
#endif
}
__device__ __forceinline__ void async_wait_le0() {
#if USE_ASYNC_LDS
    asm volatile("s_wait_asynccnt 0" ::: "memory");
#endif
}

// =====================================================================
// Kernel 1: token embedding + sinusoidal positional encoding
// =====================================================================
__global__ __launch_bounds__(256) void embed_kernel(const int* __restrict__ x,
                                                    const float* __restrict__ emb,
                                                    float* __restrict__ h) {
    const int m   = blockIdx.x;
    const int t   = m % T_;
    const int tok = x[m];
    const float neg_ln_w = logf(10000.0f) / (float)D_;
    for (int d = threadIdx.x; d < D_; d += 256) {
        int   j   = d & ~1;
        float w   = __expf(-(float)j * neg_ln_w);
        float ang = (float)t * w;
        float pe  = (d & 1) ? __cosf(ang) : __sinf(ang);
        h[(long long)m * D_ + d] = emb[(long long)tok * D_ + d] + pe;
    }
}

// =====================================================================
// Kernel 2: WMMA GEMM, double-buffered async staging, LDS-coalesced epilogue.
//   C[M,N] = epi(A[M,K] @ W[K,N]); W head-blocked: (k,n) at
//   W[(n/nb)*bstride + k*ldb + (n%nb)].
// block 256 = 8 wave32; BM=64 BN=128 BK=32; each wave: 2x2 16x16 WMMA tiles.
// Every thread issues exactly 6 async b128 copies per tile -> counted
// s_wait_asynccnt 6 overlaps tile kt+1 copies with tile kt compute.
// =====================================================================
#define BM 64
#define BN 128
#define BK 32
#define ASZ (BM * (BK + 4))        // 2304 floats, A tile with pad
#define BSZ (BK * (BN + 4))        // 4224 floats, B tile with pad
#define BUFSZ (ASZ + BSZ)
#define SMEM_FLOATS (2 * BUFSZ)

__device__ __forceinline__ void stage_tile(const float* __restrict__ A, int lda,
                                           const float* __restrict__ W, int ldb,
                                           int nb, long long bstride,
                                           int bm, int bn, int k0,
                                           float* As, float* Bs, int tid)
{
    // A tile 64x32: 512 float4 -> 2 per thread
    for (int ch = tid; ch < (BM * BK / 4); ch += 256) {
        int r = ch >> 3, c4 = (ch & 7) << 2;
        g2l_b128(&As[r * (BK + 4) + c4], A + (long long)(bm + r) * lda + (k0 + c4));
    }
    // B tile 32x128: 1024 float4 -> 4 per thread (non-temporal: weights)
    for (int ch = tid; ch < (BK * BN / 4); ch += 256) {
        int r = ch >> 5, c4 = (ch & 31) << 2;
        int gn = bn + c4;
        g2l_b128_nt(&Bs[r * (BN + 4) + c4],
                    W + (long long)(gn / nb) * bstride + (long long)(k0 + r) * ldb + (gn % nb));
    }
}

template <int RELU, int RES, int BIAS>
__global__ __launch_bounds__(256) void wmma_gemm(
    const float* __restrict__ A, int lda,
    const float* __restrict__ W, int ldb, int nb, long long bstride,
    const float* __restrict__ bias,
    const float* __restrict__ resid,
    float* __restrict__ C, int ldo,
    int M, int N, int K)
{
    __shared__ float smem[SMEM_FLOATS];

    const int tid  = threadIdx.x;
    const int lane = tid & 31;
    const int wv   = tid >> 5;
    const int wr   = wv >> 2;          // 0..1
    const int wc   = wv & 3;           // 0..3
    const int hi   = lane >> 4;
    const int n16  = lane & 15;

    const int bm = blockIdx.x * BM;
    const int bn = blockIdx.y * BN;

    v8f acc[2][2];
    acc[0][0] = vzero8(); acc[0][1] = vzero8();
    acc[1][0] = vzero8(); acc[1][1] = vzero8();

    const int nk = K / BK;
    // prologue: stage tile 0 into buffer 0
    stage_tile(A, lda, W, ldb, nb, bstride, bm, bn, 0, smem, smem + ASZ, tid);

    for (int kt = 0; kt < nk; ++kt) {
        const int cb = kt & 1;
        float* Asc = smem + cb * BUFSZ;          // current buffer (runtime addr math,
        float* Bsc = Asc + ASZ;                  //  no constant LDS pointer aggregates)
        float* Asn = smem + (cb ^ 1) * BUFSZ;    // next buffer
        float* Bsn = Asn + ASZ;

        __syncthreads();  // next buffer free (everyone done computing tile kt-1)
        if (kt + 1 < nk) {
            stage_tile(A, lda, W, ldb, nb, bstride, bm, bn, (kt + 1) * BK,
                       Asn, Bsn, tid);
            async_wait_le6();          // older 6 copies (tile kt) have landed
        } else {
            async_wait_le0();
        }
        __syncthreads();  // tile kt visible to all waves

        #pragma unroll
        for (int i = 0; i < 2; ++i) {
            // A fragment: row = lane%16; halves 0..7 -> K base hi*8,
            // halves 8..15 -> K base 16+hi*8 (ISA 7.12.2)
            v16h af;
            const float* pr = &Asc[(wr * 32 + i * 16 + n16) * (BK + 4)];
            #pragma unroll
            for (int j = 0; j < 8; ++j) af[j]     = (_Float16)pr[hi * 8 + j];
            #pragma unroll
            for (int j = 0; j < 8; ++j) af[8 + j] = (_Float16)pr[16 + hi * 8 + j];

            #pragma unroll
            for (int jj = 0; jj < 2; ++jj) {
                v16h bf;  // B fragment: col = lane%16, K = hi*8 + j (j<8), 16 + hi*8 + (j-8) (j>=8)
                int nc = wc * 32 + jj * 16 + n16;
                #pragma unroll
                for (int j = 0; j < 16; ++j)
                    bf[j] = (_Float16)Bsc[(j < 8 ? hi * 8 + j : 8 + hi * 8 + j) * (BN + 4) + nc];
                acc[i][jj] = __builtin_amdgcn_wmma_f32_16x16x32_f16(
                    false, af, false, bf, (short)0, acc[i][jj], false, false);
            }
        }
    }

    // ---- epilogue: spill accumulators through LDS for coalesced stores ----
    float* Cs = smem;                  // 64 x (BN+4) fp32 = 33 KB, fits
    __syncthreads();
    #pragma unroll
    for (int i = 0; i < 2; ++i) {
        #pragma unroll
        for (int jj = 0; jj < 2; ++jj) {
            int nloc = wc * 32 + jj * 16 + n16;
            float bv = BIAS ? bias[bn + nloc] : 0.f;
            #pragma unroll
            for (int e = 0; e < 8; ++e) {
                int   mloc = wr * 32 + i * 16 + hi * 8 + e;
                float v = acc[i][jj][e] + bv;
                if (RELU) v = fmaxf(v, 0.f);
                Cs[mloc * (BN + 4) + nloc] = v;
            }
        }
    }
    __syncthreads();
    // 64 rows x 32 float4 = 2048 chunks -> 8 per thread, fully coalesced b128
    for (int ch = tid; ch < (BM * BN / 4); ch += 256) {
        int r = ch >> 5, c4 = (ch & 31) << 2;
        float4 v = *(float4*)&Cs[r * (BN + 4) + c4];
        long long row = bm + r;
        int gn = bn + c4;
        if (RES) {
            float4 rs = *(const float4*)&resid[row * ldo + gn];
            v.x += rs.x; v.y += rs.y; v.z += rs.z; v.w += rs.w;
        }
        *(float4*)&C[row * ldo + gn] = v;
    }
}

// =====================================================================
// Kernel 3: fused attention for one (batch, head): WMMA for S=QK^T and
// O=PV, wave32 16-lane shfl_xor softmax, P relaid via LDS.
// =====================================================================
#define KS_S 68
#define PW_S 132
#define ATTN_LDS_HALVES (2 * 128 * KS_S + 4 * 16 * KS_S + 4 * 16 * PW_S)
#define ATTN_LDS_BYTES  (ATTN_LDS_HALVES * 2)

__global__ __launch_bounds__(128) void attn_kernel(const float* __restrict__ Qg,
                                                   const float* __restrict__ Kg,
                                                   const float* __restrict__ Vg,
                                                   float* __restrict__ Og,
                                                   int causal)
{
    extern __shared__ char smemraw[];
    _Float16* Ks = (_Float16*)smemraw;        // [128][KS_S]
    _Float16* Vs = Ks + 128 * KS_S;           // [128][KS_S]
    _Float16* Qw = Vs + 128 * KS_S;           // per-wave [16][KS_S]
    _Float16* Pw = Qw + 4 * 16 * KS_S;        // per-wave [16][PW_S]

    const int bh = blockIdx.x;
    const int b  = bh >> 3;
    const int h  = bh & 7;

    const float* Qp = Qg + (long long)b * T_ * HD_ + h * DH_;
    const float* Kp = Kg + (long long)b * T_ * HD_ + h * DH_;
    const float* Vp = Vg + (long long)b * T_ * HD_ + h * DH_;
    float*       Op = Og + (long long)b * T_ * HD_ + h * DH_;

    const int tid  = threadIdx.x;
    const int lane = tid & 31;
    const int wv   = tid >> 5;
    const int hi   = lane >> 4;
    const int n16  = lane & 15;

    for (int idx = tid; idx < 128 * DH_; idx += 128) {
        int r = idx >> 6, c = idx & 63;
        float kv = (r < TE_) ? Kp[(long long)r * HD_ + c] : 0.f;
        float vv = (r < TE_) ? Vp[(long long)r * HD_ + c] : 0.f;
        Ks[r * KS_S + c] = (_Float16)kv;
        Vs[r * KS_S + c] = (_Float16)vv;
    }
    __syncthreads();

    _Float16* Qt = Qw + wv * 16 * KS_S;
    _Float16* Pt = Pw + wv * 16 * PW_S;

    for (int rt = wv; rt < 8; rt += 4) {      // uniform: 2 iterations per wave
        for (int idx = lane; idx < 16 * DH_; idx += 32) {
            int r = idx >> 6, c = idx & 63;
            int m = rt * 16 + r;
            Qt[r * KS_S + c] = (m < T_) ? (_Float16)Qp[(long long)m * HD_ + c]
                                        : (_Float16)0.f;
        }
        __syncthreads();

        v8f sc[8];
        #pragma unroll
        for (int c = 0; c < 8; ++c) sc[c] = vzero8();
        #pragma unroll
        for (int e0 = 0; e0 < DH_; e0 += 32) {
            v16h af;
            const _Float16* qr = Qt + n16 * KS_S;
            #pragma unroll
            for (int j = 0; j < 8; ++j) af[j]     = qr[e0 + hi * 8 + j];
            #pragma unroll
            for (int j = 0; j < 8; ++j) af[8 + j] = qr[e0 + 16 + hi * 8 + j];
            #pragma unroll
            for (int c = 0; c < 8; ++c) {
                v16h bf;
                const _Float16* kr = Ks + (c * 16 + n16) * KS_S;
                #pragma unroll
                for (int j = 0; j < 16; ++j) bf[j] = kr[e0 + (j < 8 ? hi * 8 + j : 8 + hi * 8 + j)];
                sc[c] = __builtin_amdgcn_wmma_f32_16x16x32_f16(
                    false, af, false, bf, (short)0, sc[c], false, false);
            }
        }

        float rinv[8];
        #pragma unroll
        for (int e = 0; e < 8; ++e) {
            int   m  = rt * 16 + hi * 8 + e;
            float mx = -1e30f;
            #pragma unroll
            for (int c = 0; c < 8; ++c) {
                int   n = c * 16 + n16;
                float s = sc[c][e] * 0.125f;
                bool ok = (n < TE_) && (!causal || n <= m);
                s = ok ? s : -1e30f;
                sc[c][e] = s;
                mx = fmaxf(mx, s);
            }
            #pragma unroll
            for (int off = 1; off < 16; off <<= 1)
                mx = fmaxf(mx, __shfl_xor(mx, off, 32));
            float sum = 0.f;
            #pragma unroll
            for (int c = 0; c < 8; ++c) {
                float p = __expf(sc[c][e] - mx);
                sc[c][e] = p;
                sum += p;
            }
            #pragma unroll
            for (int off = 1; off < 16; off <<= 1)
                sum += __shfl_xor(sum, off, 32);
            rinv[e] = (sum > 0.f) ? 1.f / sum : 0.f;
        }

        #pragma unroll
        for (int c = 0; c < 8; ++c)
            #pragma unroll
            for (int e = 0; e < 8; ++e)
                Pt[(hi * 8 + e) * PW_S + c * 16 + n16] = (_Float16)sc[c][e];
        __syncthreads();

        v8f oc[4];
        #pragma unroll
        for (int c = 0; c < 4; ++c) oc[c] = vzero8();
        #pragma unroll
        for (int s0 = 0; s0 < 128; s0 += 32) {
            v16h af;
            const _Float16* pr = Pt + n16 * PW_S;
            #pragma unroll
            for (int j = 0; j < 8; ++j) af[j]     = pr[s0 + hi * 8 + j];
            #pragma unroll
            for (int j = 0; j < 8; ++j) af[8 + j] = pr[s0 + 16 + hi * 8 + j];
            #pragma unroll
            for (int c = 0; c < 4; ++c) {
                v16h bf;
                #pragma unroll
                for (int j = 0; j < 16; ++j)
                    bf[j] = Vs[(s0 + (j < 8 ? hi * 8 + j : 8 + hi * 8 + j)) * KS_S + c * 16 + n16];
                oc[c] = __builtin_amdgcn_wmma_f32_16x16x32_f16(
                    false, af, false, bf, (short)0, oc[c], false, false);
            }
        }

        #pragma unroll
        for (int e = 0; e < 8; ++e) {
            int m = rt * 16 + hi * 8 + e;
            if (m < T_) {
                #pragma unroll
                for (int c = 0; c < 4; ++c)
                    Op[(long long)m * HD_ + c * 16 + n16] = oc[c][e] * rinv[e];
            }
        }
        __syncthreads();
    }
}

// =====================================================================
// Host side
// =====================================================================
static inline void launch_gemm(const float* A, int lda,
                               const float* W, int ldb, int nb, long long bstride,
                               const float* bias, const float* resid,
                               float* C, int ldo, int M, int N, int K, int relu,
                               hipStream_t s)
{
    dim3 grid(M / BM, N / BN);
    const int has_b = bias  != nullptr;
    const int has_r = resid != nullptr;
    if (relu) {
        if (has_b) wmma_gemm<1,0,1><<<grid,256,0,s>>>(A,lda,W,ldb,nb,bstride,bias,resid,C,ldo,M,N,K);
        else       wmma_gemm<1,0,0><<<grid,256,0,s>>>(A,lda,W,ldb,nb,bstride,bias,resid,C,ldo,M,N,K);
    } else if (has_r) {
        if (has_b) wmma_gemm<0,1,1><<<grid,256,0,s>>>(A,lda,W,ldb,nb,bstride,bias,resid,C,ldo,M,N,K);
        else       wmma_gemm<0,1,0><<<grid,256,0,s>>>(A,lda,W,ldb,nb,bstride,bias,resid,C,ldo,M,N,K);
    } else {
        if (has_b) wmma_gemm<0,0,1><<<grid,256,0,s>>>(A,lda,W,ldb,nb,bstride,bias,resid,C,ldo,M,N,K);
        else       wmma_gemm<0,0,0><<<grid,256,0,s>>>(A,lda,W,ldb,nb,bstride,bias,resid,C,ldo,M,N,K);
    }
}

extern "C" void kernel_launch(void* const* d_in, const int* in_sizes, int n_in,
                              void* d_out, int out_size, void* d_ws, size_t ws_size,
                              hipStream_t stream)
{
    const int*   x    = (const int*)  d_in[0];
    const float* enc  = (const float*)d_in[1];   // [B, TE, D] = [3200, 1024]
    const float* emb  = (const float*)d_in[2];
    const float* Wq   = (const float*)d_in[3];   // [L, H, D, DH]
    const float* Wk   = (const float*)d_in[4];
    const float* Wv   = (const float*)d_in[5];
    const float* Wo   = (const float*)d_in[6];   // [L, H*DH, D]
    const float* w1   = (const float*)d_in[7];   // [L, D, 4D]
    const float* b1   = (const float*)d_in[8];
    const float* w2   = (const float*)d_in[9];   // [L, 4D, D]
    const float* b2   = (const float*)d_in[10];
    const float* Wout = (const float*)d_in[11];  // [D, V]
    const float* bout = (const float*)d_in[12];
    float* out = (float*)d_out;

    float* hA = (float*)d_ws;
    float* hB = hA + (size_t)M_ * D_;
    float* Qb = hB + (size_t)M_ * D_;
    float* Kb = Qb + (size_t)M_ * HD_;
    float* Vb = Kb + (size_t)M_ * HD_;
    float* Ob = Vb + (size_t)M_ * HD_;
    float* FF = Ob + (size_t)M_ * HD_;           // [M, 4096]

    embed_kernel<<<M_, 256, 0, stream>>>(x, emb, hA);

    float* cur = hA;
    float* tmp = hB;
    const long long headW = (long long)D_ * DH_;

    for (int i = 0; i < L_; ++i) {
        const float* Wqi = Wq + (size_t)i * H_ * D_ * DH_;
        const float* Wki = Wk + (size_t)i * H_ * D_ * DH_;
        const float* Wvi = Wv + (size_t)i * H_ * D_ * DH_;
        const float* Woi = Wo + (size_t)i * HD_ * D_;
        const float* w1i = w1 + (size_t)i * D_ * FFD_;
        const float* b1i = b1 + (size_t)i * FFD_;
        const float* w2i = w2 + (size_t)i * FFD_ * D_;
        const float* b2i = b2 + (size_t)i * D_;

        // ---- self-attention (causal) ----
        launch_gemm(cur, D_, Wqi, DH_, DH_, headW, nullptr, nullptr, Qb, HD_, M_, HD_, D_, 0, stream);
        launch_gemm(cur, D_, Wki, DH_, DH_, headW, nullptr, nullptr, Kb, HD_, M_, HD_, D_, 0, stream);
        launch_gemm(cur, D_, Wvi, DH_, DH_, headW, nullptr, nullptr, Vb, HD_, M_, HD_, D_, 0, stream);
        attn_kernel<<<B_ * H_, 128, ATTN_LDS_BYTES, stream>>>(Qb, Kb, Vb, Ob, 1);
        launch_gemm(Ob, HD_, Woi, D_, D_, 0, nullptr, cur, tmp, D_, M_, D_, HD_, 0, stream); // h2 = O@Wo + h

        // ---- cross-attention (same weights; K/V from encoder state) ----
        launch_gemm(tmp, D_, Wqi, DH_, DH_, headW, nullptr, nullptr, Qb, HD_, M_, HD_, D_, 0, stream);
        launch_gemm(enc, D_, Wki, DH_, DH_, headW, nullptr, nullptr, Kb, HD_, M_, HD_, D_, 0, stream);
        launch_gemm(enc, D_, Wvi, DH_, DH_, headW, nullptr, nullptr, Vb, HD_, M_, HD_, D_, 0, stream);
        attn_kernel<<<B_ * H_, 128, ATTN_LDS_BYTES, stream>>>(Qb, Kb, Vb, Ob, 0);
        launch_gemm(Ob, HD_, Woi, D_, D_, 0, nullptr, tmp, cur, D_, M_, D_, HD_, 0, stream); // h3 = O@Wo + h2

        // ---- feed-forward ----
        launch_gemm(cur, D_, w1i, FFD_, FFD_, 0, b1i, nullptr, FF, FFD_, M_, FFD_, D_, 1, stream);
        launch_gemm(FF, FFD_, w2i, D_, D_, 0, b2i, cur, tmp, D_, M_, D_, FFD_, 0, stream);   // h4 = ff + h3

        float* sw = cur; cur = tmp; tmp = sw;
    }

    launch_gemm(cur, D_, Wout, V_, V_, 0, bout, nullptr, out, V_, M_, V_, D_, 0, stream);
}
